// GCNLayer_55748675502408
// MI455X (gfx1250) — matrix operand order repaired and checked by reading, not compiled, not run on hardware
//
#include <hip/hip_runtime.h>
#include <stdint.h>

#define N_NODES 50000
#define N_EDGES 800000
#define D_MODEL 128
#define BN_EPS 1e-5f

typedef __bf16 bf16_t;
typedef __bf16 v16bf __attribute__((ext_vector_type(16)));
typedef float  v8f   __attribute__((ext_vector_type(8)));

// Device-scope relaxed f32 atomic add -> single global_atomic_add_f32
// (non-returning, no CAS loop, no system-scope fences).
__device__ __forceinline__ void atomic_add_f32(float* p, float v) {
    __hip_atomic_fetch_add(p, v, __ATOMIC_RELAXED, __HIP_MEMORY_SCOPE_AGENT);
}

// ---------------- zero-init workspace regions ----------------
__global__ __launch_bounds__(256) void k_zero(float* __restrict__ p, long n) {
    long i = (long)blockIdx.x * blockDim.x + threadIdx.x;
    long stride = (long)gridDim.x * blockDim.x;
    for (; i < n; i += stride) p[i] = 0.0f;
}

// ---------------- pre-swizzle weights into WMMA B-fragment layout (bf16) ----
// For V_WMMA_F32_16X16X32_BF16, B (32x16, KxN) per-lane layout:
//   lane element i (0..7): K = 8*(lane>=16) + i; i (8..15): K = 16 + 8*(lane>=16) + (i-8); N = lane&15.
// We store, for (coltile ct 0..7, kblock kb 0..3, lane 0..31), the 16 bf16
// values contiguously so the GEMM kernel does one aligned 32B load per frag.
__global__ __launch_bounds__(256) void k_prep_w(const float* __restrict__ Wg,
                                                const float* __restrict__ Wl,
                                                bf16_t* __restrict__ Wg_sw,
                                                bf16_t* __restrict__ Wl_sw) {
    int idx = blockIdx.x * 256 + threadIdx.x;
    if (idx >= D_MODEL * D_MODEL) return;
    int i    = idx & 15;
    int g    = idx >> 4;         // fragment group 0..1023
    int lane = g & 31;
    int kb   = (g >> 5) & 3;
    int ct   = g >> 7;           // 0..7
    int hi   = lane >> 4;
    int n    = ct * 16 + (lane & 15);
    int k    = kb * 32 + (i < 8 ? hi * 8 + i : 16 + hi * 8 + (i - 8));
    Wg_sw[idx] = (bf16_t)Wg[k * D_MODEL + n];
    Wl_sw[idx] = (bf16_t)Wl[k * D_MODEL + n];
}

// ---------------- degree counts ----------------
__global__ __launch_bounds__(256) void k_degrees(const long long* __restrict__ src,
                                                 const long long* __restrict__ dst,
                                                 float* __restrict__ deg_out,
                                                 float* __restrict__ deg_in) {
    int e = blockIdx.x * 256 + threadIdx.x;
    if (e >= N_EDGES) return;
    atomic_add_f32(&deg_out[(int)src[e]], 1.0f);
    atomic_add_f32(&deg_in [(int)dst[e]], 1.0f);
}

// deg -> rsqrt(max(deg,1)) in place
__global__ __launch_bounds__(256) void k_norm(float* __restrict__ deg_out,
                                              float* __restrict__ deg_in) {
    int i = blockIdx.x * 256 + threadIdx.x;
    if (i >= N_NODES) return;
    deg_out[i] = rsqrtf(fmaxf(deg_out[i], 1.0f));
    deg_in[i]  = rsqrtf(fmaxf(deg_in[i],  1.0f));
}

// ---------------- edge gather-scatter: agg[dst] += x[src]*norm_src[src] -----
// one wave32 per edge, each lane handles 4 consecutive features (float4 load,
// 4 device-scope relaxed f32 atomics). x and agg are L2-resident (25.6MB each),
// so this phase runs at L2 atomic throughput, not HBM bandwidth.
__global__ __launch_bounds__(256) void k_scatter(const long long* __restrict__ src,
                                                 const long long* __restrict__ dst,
                                                 const float* __restrict__ x,
                                                 const float* __restrict__ norm_src,
                                                 float* __restrict__ agg) {
    long t = (long)blockIdx.x * 256 + threadIdx.x;
    int e = (int)(t >> 5);
    int lane = threadIdx.x & 31;
    if (e >= N_EDGES) return;
    int s = (int)src[e];
    int d = (int)dst[e];
    float ns = norm_src[s];
    const float4 v = ((const float4*)(x + (size_t)s * D_MODEL))[lane];
    float* ad = agg + (size_t)d * D_MODEL + 4 * lane;
    atomic_add_f32(ad + 0, v.x * ns);
    atomic_add_f32(ad + 1, v.y * ns);
    atomic_add_f32(ad + 2, v.z * ns);
    atomic_add_f32(ad + 3, v.w * ns);
}

// ---------------- fused GEMM: t = (agg*norm_dst)@Wg + x@Wl + x + bg + bl ----
// 8 waves/block; block owns a 16-row node tile (bf16 in LDS), each wave owns
// one 16-col output tile; 4 K-steps x 2 matrices = 8 v_wmma_f32_16x16x32_bf16
// with fp32 accumulators seeded with the residual + biases. In-place over agg.
__global__ __launch_bounds__(256) void k_gemm_fused(const float* aggin,
                                                    const float* __restrict__ x,
                                                    const float* __restrict__ norm_dst,
                                                    const bf16_t* __restrict__ Wg_sw,
                                                    const bf16_t* __restrict__ Wl_sw,
                                                    const float* __restrict__ b_gcn,
                                                    const float* __restrict__ b_lin,
                                                    float* t_out) {
    __shared__ bf16_t a1s[16 * D_MODEL];   // (agg * norm_dst) tile
    __shared__ bf16_t a2s[16 * D_MODEL];   // x tile
    const int row0 = blockIdx.x * 16;
    const int tid  = threadIdx.x;

    for (int i = tid; i < 16 * D_MODEL; i += 256) {
        int r = i >> 7;
        int gidx = row0 + r;
        int cc = i & 127;
        a1s[i] = (bf16_t)(aggin[(size_t)gidx * D_MODEL + cc] * norm_dst[gidx]);
        a2s[i] = (bf16_t)(x[(size_t)gidx * D_MODEL + cc]);
    }
    __syncthreads();

    const int wave = tid >> 5;       // 0..7 -> column tile
    const int lane = tid & 31;
    const int m    = lane & 15;
    const int hi   = lane >> 4;
    const int col  = wave * 16 + m;

    // C/D 16x16 f32 layout: lanes 0-15 N=0..15 / M=r ; lanes 16-31 M=8+r
    v8f c;
    float bias = b_gcn[col] + b_lin[col];
#pragma unroll
    for (int r = 0; r < 8; ++r)
        c[r] = x[(size_t)(row0 + r + 8 * hi) * D_MODEL + col] + bias;

#pragma unroll
    for (int kb = 0; kb < 4; ++kb) {
        // A (16x32 bf16) per-lane: i in 0..7 -> K = kb*32 + 8*hi + i,
        //                          i in 8..15 -> K = kb*32 + 16 + 8*hi + (i-8)
        int kbase = kb * 32 + hi * 8;
        v16bf a1, a2;
#pragma unroll
        for (int i = 0; i < 8; ++i) {
            a1[i]     = a1s[m * D_MODEL + kbase + i];
            a1[8 + i] = a1s[m * D_MODEL + kbase + 16 + i];
            a2[i]     = a2s[m * D_MODEL + kbase + i];
            a2[8 + i] = a2s[m * D_MODEL + kbase + 16 + i];
        }
        const v16bf b1 = *(const v16bf*)&Wg_sw[((wave * 4 + kb) * 32 + lane) * 16];
        const v16bf b2 = *(const v16bf*)&Wl_sw[((wave * 4 + kb) * 32 + lane) * 16];
        c = __builtin_amdgcn_wmma_f32_16x16x32_bf16(false, a1, false, b1,
                                                    (short)0, c, false, false);
        c = __builtin_amdgcn_wmma_f32_16x16x32_bf16(false, a2, false, b2,
                                                    (short)0, c, false, false);
    }

#pragma unroll
    for (int r = 0; r < 8; ++r)
        t_out[(size_t)(row0 + r + 8 * hi) * D_MODEL + col] = c[r];
}

// ---------------- BatchNorm column statistics ----------------
__global__ __launch_bounds__(128) void k_bn_stats(const float* __restrict__ t,
                                                  float* __restrict__ sums,
                                                  float* __restrict__ sumsq) {
    int c = threadIdx.x;                 // 0..127, coalesced across lanes
    float s = 0.0f, s2 = 0.0f;
    for (int r = blockIdx.x; r < N_NODES; r += gridDim.x) {
        float v = t[(size_t)r * D_MODEL + c];
        s += v; s2 += v * v;
    }
    atomic_add_f32(&sums[c], s);
    atomic_add_f32(&sumsq[c], s2);
}

// ---------------- BN apply + ReLU ----------------
__global__ __launch_bounds__(256) void k_bn_apply(const float* __restrict__ t,
                                                  const float* __restrict__ sums,
                                                  const float* __restrict__ sumsq,
                                                  const float* __restrict__ gamma,
                                                  const float* __restrict__ beta,
                                                  float* __restrict__ out) {
    long i = (long)blockIdx.x * 256 + threadIdx.x;
    const long total = (long)N_NODES * D_MODEL;
    if (i >= total) return;
    int c = (int)(i & 127);
    const float invN = 1.0f / (float)N_NODES;
    float mean = sums[c] * invN;
    float var  = sumsq[c] * invN - mean * mean;
    float inv  = rsqrtf(var + BN_EPS);
    float v = (t[i] - mean) * inv * gamma[c] + beta[c];
    out[i] = fmaxf(v, 0.0f);
}

// ---------------- launcher ----------------
extern "C" void kernel_launch(void* const* d_in, const int* in_sizes, int n_in,
                              void* d_out, int out_size, void* d_ws, size_t ws_size,
                              hipStream_t stream) {
    const float*     x     = (const float*)d_in[0];
    const float*     Wg    = (const float*)d_in[1];
    const float*     bg    = (const float*)d_in[2];
    const float*     Wl    = (const float*)d_in[3];
    const float*     bl    = (const float*)d_in[4];
    const float*     gamma = (const float*)d_in[5];
    const float*     beta  = (const float*)d_in[6];
    const long long* src   = (const long long*)d_in[7];   // int64 per reference
    const long long* dst   = (const long long*)d_in[8];
    float* out = (float*)d_out;

    // workspace layout (floats)
    float* wsf   = (float*)d_ws;
    float* agg   = wsf;                                   // N*D (reused as t)
    float* dego  = wsf + (size_t)N_NODES * D_MODEL;       // N (-> norm_src)
    float* degi  = dego + N_NODES;                        // N (-> norm_dst)
    float* sums  = degi + N_NODES;                        // 128
    float* sumsq = sums + D_MODEL;                        // 128
    const long zcount = 6500352;                          // padded float count
    bf16_t* Wg_sw = (bf16_t*)(wsf + zcount);              // 16384 bf16
    bf16_t* Wl_sw = Wg_sw + D_MODEL * D_MODEL;            // 16384 bf16

    k_zero<<<2048, 256, 0, stream>>>(wsf, zcount);
    k_prep_w<<<(D_MODEL * D_MODEL + 255) / 256, 256, 0, stream>>>(Wg, Wl, Wg_sw, Wl_sw);
    k_degrees<<<(N_EDGES + 255) / 256, 256, 0, stream>>>(src, dst, dego, degi);
    k_norm<<<(N_NODES + 255) / 256, 256, 0, stream>>>(dego, degi);
    k_scatter<<<((long)N_EDGES * 32 + 255) / 256, 256, 0, stream>>>(src, dst, x, dego, agg);
    k_gemm_fused<<<N_NODES / 16, 256, 0, stream>>>(agg, x, degi, Wg_sw, Wl_sw, bg, bl, agg);
    k_bn_stats<<<512, 128, 0, stream>>>(agg, sums, sumsq);
    k_bn_apply<<<((long)N_NODES * D_MODEL + 255) / 256, 256, 0, stream>>>(agg, sums, sumsq, gamma, beta, out);
}
